// HybridQGNN_65936337928328
// MI455X (gfx1250) — compile-verified, matrix operand order checked by reading, not run on hardware
//
#include <hip/hip_runtime.h>
#include <hip/hip_bf16.h>
#include <math.h>

typedef __bf16 bf16_t;
typedef __attribute__((ext_vector_type(16))) __bf16 v16bf;
typedef __attribute__((ext_vector_type(8)))  float  v8f;

#define N_NODES 32768
#define N_EDGES 131072
#define FIN     128
#define HID     256
#define EMBD    128
#define NB      1024
#define NQ      10
#define NL      4
#define PI_F    3.14159f

static inline int cdiv(int a, int b) { return (a + b - 1) / b; }

// ---------- f32 -> bf16 conversion ----------
__global__ void f32_to_bf16_kernel(const float* __restrict__ in, bf16_t* __restrict__ out, int n) {
    int i = blockIdx.x * blockDim.x + threadIdx.x;
    if (i < n) out[i] = (bf16_t)in[i];
}

// ---------- pack weights (KxN f32, row-major) into WMMA fragment-major bf16 ----------
// Layout: [panel p = k/32][subtile s = col/16][lane 0..31][elem 0..15], so a 64-column
// K-step panel is 4 KB of *contiguous* memory -> async-copied straight into LDS.
// Fragment semantics (ISA 16-bit B layout): lane L holds col = s*16 + (L&15);
// elem e holds k = 32p + 16*(e>>3) + 8*(L>>4) + 2*((e>>1)&3) + (e&1).
__global__ void pack_w_kernel(const float* __restrict__ W, bf16_t* __restrict__ Wpk, int K, int N) {
    int idx = blockIdx.x * blockDim.x + threadIdx.x;
    if (idx >= K * N) return;
    int perPanel = (N >> 4) << 9;           // (N/16)*512 elems per K-step panel
    int p    = idx / perPanel;
    int rem  = idx % perPanel;
    int s    = rem >> 9;
    int r2   = rem & 511;
    int lane = r2 >> 4, e = r2 & 15;
    int col  = (s << 4) + (lane & 15);
    int kloc = ((e >> 3) << 4) + ((lane >> 4) << 3) + (((e >> 1) & 3) << 1) + (e & 1);
    Wpk[idx] = (bf16_t)W[(size_t)((p << 5) + kloc) * N + col];
}

// ---------- WMMA GEMM: C(MxN,f32) = A(MxK,bf16 row-major) * Bpk(packed bf16) ----------
// Block = 8 waves -> 128x64 C macro-tile; wave w owns rows [bm+16w,+16), 4 col subtiles.
// B panels are double-buffered in LDS via global_load_async_to_lds_b128 (ASYNCcnt),
// so panel p+1 streams in while panel p feeds 4 WMMAs per wave.
__global__ void __launch_bounds__(256)
wmma_gemm_bf16(const bf16_t* __restrict__ A, const bf16_t* __restrict__ Bpk,
               float* __restrict__ C, int M, int N, int K) {
    __shared__ bf16_t lds_fr[2][2048];       // 2 x 4KB panel buffers, fragment order

    int t      = threadIdx.x;
    int waveId = t >> 5;
    int lane   = t & 31;
    int half   = lane >> 4;                  // K-half selector (ISA 16-bit A layout)
    int idx16  = lane & 15;                  // A: row in tile; C: column in tile
    int tm     = blockIdx.x * 128 + waveId * 16;
    int tn     = blockIdx.y * 64;
    int steps  = K >> 5;

    // low 32 bits of the flat address of a __shared__ object = LDS byte offset (ISA 10.2)
    uint32_t lds_base = (uint32_t)(uintptr_t)&lds_fr[0][0];
    const char* gsrc   = (const char*)Bpk + ((size_t)(tn >> 4) << 10) + (size_t)t * 16;
    size_t      gstep  = (size_t)(N >> 4) << 10;   // bytes per K-step panel

    // kick off panel 0 into buffer 0 (one b128 per thread = 4KB per block)
    {
        uint32_t ldst = lds_base + t * 16;
        asm volatile("global_load_async_to_lds_b128 %0, %1, off"
                     :: "v"(ldst), "v"(gsrc) : "memory");
    }

    v8f acc0 = {}, acc1 = {}, acc2 = {}, acc3 = {};
    const bf16_t* Arow = A + (size_t)(tm + idx16) * K;

    for (int p = 0; p < steps; ++p) {
        asm volatile("s_wait_asynccnt 0x0" ::: "memory");   // my panel-p chunk landed
        __syncthreads();                                    // everyone's chunk landed; prev reads done
        if (p + 1 < steps) {                                // stream next panel into other buffer
            uint32_t ldst   = lds_base + (uint32_t)(((p + 1) & 1) << 12) + t * 16;
            const char* ga  = gsrc + (size_t)(p + 1) * gstep;
            asm volatile("global_load_async_to_lds_b128 %0, %1, off"
                         :: "v"(ldst), "v"(ga) : "memory");
        }

        int kk = p << 5;
        // A fragment: two contiguous 16B chunks per lane (compiler -> global_load_b128 x2)
        v16bf a;
#pragma unroll
        for (int q = 0; q < 8; ++q) {
            int k0 = ((q >> 2) << 4) + ((q & 3) << 1) + (half << 3);
            a[2 * q]     = Arow[kk + k0];
            a[2 * q + 1] = Arow[kk + k0 + 1];
        }
        // B fragments: 32 contiguous bytes per lane per subtile (ds_load_b128 x2 each)
        const bf16_t* fb = &lds_fr[p & 1][0];
        v16bf b0 = *(const v16bf*)&fb[0 * 512 + lane * 16];
        v16bf b1 = *(const v16bf*)&fb[1 * 512 + lane * 16];
        v16bf b2 = *(const v16bf*)&fb[2 * 512 + lane * 16];
        v16bf b3 = *(const v16bf*)&fb[3 * 512 + lane * 16];

        acc0 = __builtin_amdgcn_wmma_f32_16x16x32_bf16(false, a, false, b0, (short)0, acc0, false, false);
        acc1 = __builtin_amdgcn_wmma_f32_16x16x32_bf16(false, a, false, b1, (short)0, acc1, false, false);
        acc2 = __builtin_amdgcn_wmma_f32_16x16x32_bf16(false, a, false, b2, (short)0, acc2, false, false);
        acc3 = __builtin_amdgcn_wmma_f32_16x16x32_bf16(false, a, false, b3, (short)0, acc3, false, false);
    }

#pragma unroll
    for (int r = 0; r < 8; ++r) {            // C layout: VGPR r -> M = r + 8*half, col = idx16
        int row = tm + r + (half << 3);
        float* crow = C + (size_t)row * N + tn;
        crow[ 0 + idx16] = acc0[r];
        crow[16 + idx16] = acc1[r];
        crow[32 + idx16] = acc2[r];
        crow[48 + idx16] = acc3[r];
    }
}

// ---------- GCN edge aggregation: agg[dst] += h[src] * rsqrt(deg+1)[src]*rsqrt(deg+1)[dst] ----------
__global__ void gcn_agg_kernel(const float* __restrict__ h, const int* __restrict__ src,
                               const int* __restrict__ dst, const float* __restrict__ deg,
                               float* __restrict__ agg, int E, int H) {
    int idx    = blockIdx.x * blockDim.x + threadIdx.x;
    int chunks = H >> 2;
    if (idx >= E * chunks) return;
    int e = idx / chunks;
    int c = (idx % chunks) << 2;
    int s = src[e], d = dst[e];
    float coef = rsqrtf(deg[s] + 1.0f) * rsqrtf(deg[d] + 1.0f);
    const float4 hv = *(const float4*)(h + (size_t)s * H + c);
    float* ap = agg + (size_t)d * H + c;
    atomicAdd(ap + 0, hv.x * coef);
    atomicAdd(ap + 1, hv.y * coef);
    atomicAdd(ap + 2, hv.z * coef);
    atomicAdd(ap + 3, hv.w * coef);
}

// ---------- degree count ----------
__global__ void degree_kernel(const int* __restrict__ dst, float* __restrict__ deg, int E) {
    int e = blockIdx.x * blockDim.x + threadIdx.x;
    if (e < E) atomicAdd(&deg[dst[e]], 1.0f);
}

// ---------- conv1 combine: relu(agg + h/(deg+1) + b) -> bf16 (feeds next WMMA GEMM) ----------
__global__ void gcn_combine_bf16_kernel(const float* __restrict__ agg, const float* __restrict__ h,
                                        const float* __restrict__ deg, const float* __restrict__ bias,
                                        bf16_t* __restrict__ out, int n, int H) {
    int i = blockIdx.x * blockDim.x + threadIdx.x;
    if (i >= n * H) return;
    int node = i / H, d = i % H;
    float v = agg[i] + h[i] / (deg[node] + 1.0f) + bias[d];
    out[i] = (bf16_t)fmaxf(v, 0.0f);
}

// ---------- conv2 combine fused with segment-mean accumulation ----------
__global__ void gcn_combine2_seg_kernel(const float* __restrict__ agg, const float* __restrict__ h,
                                        const float* __restrict__ deg, const float* __restrict__ bias,
                                        const int* __restrict__ batch, float* __restrict__ tot,
                                        float* __restrict__ cnt, int n, int H) {
    int i = blockIdx.x * blockDim.x + threadIdx.x;
    if (i >= n * H) return;
    int node = i / H, d = i % H;
    float v = fmaxf(agg[i] + h[i] / (deg[node] + 1.0f) + bias[d], 0.0f);
    int b = batch[node];
    atomicAdd(&tot[(size_t)b * H + d], v);
    if (d == 0) atomicAdd(&cnt[b], 1.0f);
}

__global__ void seg_fin_kernel(const float* __restrict__ tot, const float* __restrict__ cnt,
                               float* __restrict__ out, int nb, int H) {
    int i = blockIdx.x * blockDim.x + threadIdx.x;
    if (i >= nb * H) return;
    out[i] = tot[i] / fmaxf(cnt[i / H], 1.0f);
}

// ---------- q_in = tanh(emb @ projW + projb) * PI ----------
__global__ void qin_kernel(const float* __restrict__ emb, const float* __restrict__ projW,
                           const float* __restrict__ projb, float* __restrict__ qin) {
    int i = blockIdx.x * blockDim.x + threadIdx.x;
    if (i >= NB * NQ) return;
    int b = i / NQ, q = i % NQ;
    float s = projb[q];
    const float* er = emb + (size_t)b * EMBD;
    for (int k = 0; k < EMBD; ++k) s += er[k] * projW[k * NQ + q];
    qin[i] = tanhf(s) * PI_F;
}

// ---------- edge-attr pooling by batch[src] ----------
__global__ void epool_acc_kernel(const float* __restrict__ ea, const int* __restrict__ src,
                                 const int* __restrict__ batch, float* __restrict__ tot,
                                 float* __restrict__ cnt, int E) {
    int e = blockIdx.x * blockDim.x + threadIdx.x;
    if (e >= E) return;
    int b = batch[src[e]];
    atomicAdd(&tot[b * 4 + 0], ea[e * 4 + 0]);
    atomicAdd(&tot[b * 4 + 1], ea[e * 4 + 1]);
    atomicAdd(&tot[b * 4 + 2], ea[e * 4 + 2]);
    atomicAdd(&tot[b * 4 + 3], ea[e * 4 + 3]);
    atomicAdd(&cnt[b], 1.0f);
}

__global__ void ep_kernel(const float* __restrict__ tot, const float* __restrict__ cnt,
                          const float* __restrict__ epW, const float* __restrict__ epb,
                          float* __restrict__ ep) {
    int i = blockIdx.x * blockDim.x + threadIdx.x;
    if (i >= NB * (NQ - 1)) return;
    int b = i / (NQ - 1), j = i % (NQ - 1);
    float inv = 1.0f / fmaxf(cnt[b], 1.0f);
    float s = epb[j];
    for (int k = 0; k < 4; ++k) s += tot[b * 4 + k] * inv * epW[k * (NQ - 1) + j];
    ep[i] = s;
}

// ---------- VQC: one workgroup per graph, 1024-dim real state in LDS ----------
__global__ void __launch_bounds__(256)
vqc_kernel(const float* __restrict__ qin, const float* __restrict__ w,
           const float* __restrict__ ep, float* __restrict__ qout) {
    __shared__ float st[1024];
    __shared__ float red[256];
    int b = blockIdx.x, t = threadIdx.x;

    for (int i = t; i < 1024; i += 256) st[i] = (i == 0) ? 1.0f : 0.0f;
    __syncthreads();

    auto ry = [&](int wqb, float th) {
        float c = __cosf(0.5f * th), s = __sinf(0.5f * th);
        int str = 1 << (9 - wqb);
        for (int p = t; p < 512; p += 256) {
            int a = p / str, cc = p % str;
            int i0 = a * (str << 1) + cc;
            int i1 = i0 + str;
            float a0 = st[i0], a1 = st[i1];
            st[i0] = c * a0 - s * a1;
            st[i1] = s * a0 + c * a1;
        }
        __syncthreads();
    };
    auto cry = [&](int wqb, float th) {    // control wqb, target wqb+1; 256 affected pairs
        float c = __cosf(0.5f * th), s = __sinf(0.5f * th);
        int str = 1 << (8 - wqb);
        int a = t / str, cc = t % str;
        int i0 = a * (str << 2) + (str << 1) + cc;
        int i1 = i0 + str;
        float a0 = st[i0], a1 = st[i1];
        st[i0] = c * a0 - s * a1;
        st[i1] = s * a0 + c * a1;
        __syncthreads();
    };

    for (int i = 0; i < NQ; ++i) ry(i, qin[b * NQ + i]);
    for (int l = 0; l < NL; ++l) {
        for (int i = 0; i < NQ; ++i) ry(i, w[l * NQ + i]);
        for (int i = 0; i < NQ - 1; ++i) cry(i, ep[b * (NQ - 1) + i]);
    }

    float z[NQ];
#pragma unroll
    for (int q = 0; q < NQ; ++q) z[q] = 0.0f;
    for (int sIdx = t; sIdx < 1024; sIdx += 256) {
        float p = st[sIdx] * st[sIdx];
#pragma unroll
        for (int q = 0; q < NQ; ++q)
            z[q] += ((sIdx >> (9 - q)) & 1) ? -p : p;
    }
    for (int q = 0; q < NQ; ++q) {
        red[t] = z[q];
        __syncthreads();
        for (int off2 = 128; off2 > 0; off2 >>= 1) {
            if (t < off2) red[t] += red[t + off2];
            __syncthreads();
        }
        if (t == 0) qout[b * NQ + q] = red[0];
        __syncthreads();
    }
}

// ---------- classifier head: relu([emb,qout] @ cW1 + cb1) @ cW2 + cb2 ----------
__global__ void clf_kernel(const float* __restrict__ emb, const float* __restrict__ qout,
                           const float* __restrict__ cW1, const float* __restrict__ cb1,
                           const float* __restrict__ cW2, const float* __restrict__ cb2,
                           float* __restrict__ out) {
    int b = blockIdx.x * blockDim.x + threadIdx.x;
    if (b >= NB) return;
    float o = cb2[0];
    const float* er = emb + (size_t)b * EMBD;
    const float* qr = qout + (size_t)b * NQ;
    for (int j = 0; j < 16; ++j) {
        float acc = cb1[j];
        for (int k = 0; k < EMBD; ++k) acc += er[k] * cW1[k * 16 + j];
        for (int k = 0; k < NQ; ++k) acc += qr[k] * cW1[(EMBD + k) * 16 + j];
        o += fmaxf(acc, 0.0f) * cW2[j];
    }
    out[b] = o;
}

extern "C" void kernel_launch(void* const* d_in, const int* in_sizes, int n_in,
                              void* d_out, int out_size, void* d_ws, size_t ws_size,
                              hipStream_t stream) {
    const float* x     = (const float*)d_in[0];
    const int*   eidx  = (const int*)d_in[1];
    const int*   batch = (const int*)d_in[2];
    const float* eattr = (const float*)d_in[3];
    const float* W1    = (const float*)d_in[4];
    const float* b1    = (const float*)d_in[5];
    const float* W2    = (const float*)d_in[6];
    const float* b2    = (const float*)d_in[7];
    const float* projW = (const float*)d_in[8];
    const float* projb = (const float*)d_in[9];
    const float* vqcW  = (const float*)d_in[10];
    const float* epW   = (const float*)d_in[11];
    const float* epb   = (const float*)d_in[12];
    const float* cW1   = (const float*)d_in[13];
    const float* cb1   = (const float*)d_in[14];
    const float* cW2   = (const float*)d_in[15];
    const float* cb2   = (const float*)d_in[16];
    const int* src = eidx;
    const int* dst = eidx + N_EDGES;

    char*  ws  = (char*)d_ws;
    size_t off = 0;
    auto alloc = [&](size_t bytes) -> char* {
        off = (off + 255) & ~(size_t)255;
        char* p = ws + off;
        off += bytes;
        return p;
    };
    bf16_t* xb   = (bf16_t*)alloc((size_t)N_NODES * FIN * 2);
    bf16_t* w1pk = (bf16_t*)alloc((size_t)FIN * HID * 2);
    bf16_t* w2pk = (bf16_t*)alloc((size_t)HID * EMBD * 2);
    float*  h1   = (float*) alloc((size_t)N_NODES * HID * 4);   // reused as h2 (N*EMBD)
    float*  agg  = (float*) alloc((size_t)N_NODES * HID * 4);   // reused as agg2
    bf16_t* h1r  = (bf16_t*)alloc((size_t)N_NODES * HID * 2);
    float*  deg  = (float*) alloc((size_t)N_NODES * 4);
    float*  etot = (float*) alloc((size_t)NB * EMBD * 4);
    float*  ecnt = (float*) alloc((size_t)NB * 4);
    float*  emb  = (float*) alloc((size_t)NB * EMBD * 4);
    float*  qin  = (float*) alloc((size_t)NB * NQ * 4);
    float*  ptot = (float*) alloc((size_t)NB * 4 * 4);
    float*  pcnt = (float*) alloc((size_t)NB * 4);
    float*  epv  = (float*) alloc((size_t)NB * (NQ - 1) * 4);
    float*  qout = (float*) alloc((size_t)NB * NQ * 4);
    float*  h2 = h1;

    // zero accumulators (graph-capturable memset nodes)
    hipMemsetAsync(deg,  0, (size_t)N_NODES * 4, stream);
    hipMemsetAsync(agg,  0, (size_t)N_NODES * HID * 4, stream);
    hipMemsetAsync(etot, 0, (size_t)NB * EMBD * 4, stream);
    hipMemsetAsync(ecnt, 0, (size_t)NB * 4, stream);
    hipMemsetAsync(ptot, 0, (size_t)NB * 16, stream);
    hipMemsetAsync(pcnt, 0, (size_t)NB * 4, stream);

    const int TB = 256;
    // convert activations to bf16; pack weights into fragment-major bf16 panels
    f32_to_bf16_kernel<<<cdiv(N_NODES * FIN, TB), TB, 0, stream>>>(x, xb, N_NODES * FIN);
    pack_w_kernel<<<cdiv(FIN * HID, TB), TB, 0, stream>>>(W1, w1pk, FIN, HID);
    pack_w_kernel<<<cdiv(HID * EMBD, TB), TB, 0, stream>>>(W2, w2pk, HID, EMBD);

    degree_kernel<<<cdiv(N_EDGES, TB), TB, 0, stream>>>(dst, deg, N_EDGES);

    // ---- GCN conv 1: h1 = xb @ W1 ----
    {
        dim3 grid(N_NODES / 128, HID / 64);
        wmma_gemm_bf16<<<grid, TB, 0, stream>>>(xb, w1pk, h1, N_NODES, HID, FIN);
        gcn_agg_kernel<<<cdiv(N_EDGES * (HID / 4), TB), TB, 0, stream>>>(h1, src, dst, deg, agg, N_EDGES, HID);
        gcn_combine_bf16_kernel<<<cdiv(N_NODES * HID, TB), TB, 0, stream>>>(agg, h1, deg, b1, h1r, N_NODES, HID);
    }
    // ---- GCN conv 2 (h2 reuses h1 buffer, agg2 reuses agg buffer) ----
    hipMemsetAsync(agg, 0, (size_t)N_NODES * EMBD * 4, stream);
    {
        dim3 grid(N_NODES / 128, EMBD / 64);
        wmma_gemm_bf16<<<grid, TB, 0, stream>>>(h1r, w2pk, h2, N_NODES, EMBD, HID);
        gcn_agg_kernel<<<cdiv(N_EDGES * (EMBD / 4), TB), TB, 0, stream>>>(h2, src, dst, deg, agg, N_EDGES, EMBD);
        gcn_combine2_seg_kernel<<<cdiv(N_NODES * EMBD, TB), TB, 0, stream>>>(
            agg, h2, deg, b2, batch, etot, ecnt, N_NODES, EMBD);
    }
    // ---- graph pooling finalize + projection ----
    seg_fin_kernel<<<cdiv(NB * EMBD, TB), TB, 0, stream>>>(etot, ecnt, emb, NB, EMBD);
    qin_kernel<<<cdiv(NB * NQ, TB), TB, 0, stream>>>(emb, projW, projb, qin);

    // ---- edge-attr pooling + entangler params ----
    epool_acc_kernel<<<cdiv(N_EDGES, TB), TB, 0, stream>>>(eattr, src, batch, ptot, pcnt, N_EDGES);
    ep_kernel<<<cdiv(NB * (NQ - 1), TB), TB, 0, stream>>>(ptot, pcnt, epW, epb, epv);

    // ---- VQC simulation: one workgroup per graph ----
    vqc_kernel<<<NB, 256, 0, stream>>>(qin, vqcW, epv, qout);

    // ---- classifier head ----
    clf_kernel<<<cdiv(NB, TB), TB, 0, stream>>>(emb, qout, cW1, cb1, cW2, cb2, (float*)d_out);
}
